// TransformerBlock_41257455846001
// MI455X (gfx1250) — matrix operand order repaired and checked by reading, compiled, not run on hardware
//
#include <hip/hip_runtime.h>
#include <hip/hip_bf16.h>
#include <cmath>

// ---------------------------------------------------------------------------
// CDNA5 (gfx1250) bf16 WMMA transformer block.
// D = A(16x32 bf16) x B(32x16 bf16) + C(16x16 f32), wave32.
// ---------------------------------------------------------------------------

typedef __attribute__((ext_vector_type(16))) __bf16 v16bf;
typedef __attribute__((ext_vector_type(8)))  float  v8f;

union Frag { v16bf v; uint4 u[2]; };

#define T_ 4096
#define C_ 1024
#define H_ 16
#define D_ 64

static __device__ __forceinline__ v8f wmma_bf16(const Frag& a, const Frag& b, v8f c) {
  // 8 args: (neg_a, A, neg_b, B, c_mod, C, reuse_a, reuse_b)
  return __builtin_amdgcn_wmma_f32_16x16x32_bf16(false, a.v, false, b.v, (short)0, c,
                                                 false, false);
}

// ---- gfx1250 async global->LDS copy (guarded; falls back to sync copy) ----
#if defined(__gfx1250__) && __has_builtin(__builtin_amdgcn_global_load_async_to_lds_b128)
#define ASYNC_LDS 1
#endif

#ifdef ASYNC_LDS
typedef int v4i __attribute__((vector_size(16)));            // matches 'V4i'
typedef __attribute__((address_space(1))) v4i as1_v4i;       // global int4
typedef __attribute__((address_space(3))) v4i as3_v4i;       // LDS int4
#endif

static __device__ __forceinline__ void async_copy_b128(const void* g, void* l) {
#ifdef ASYNC_LDS
  __builtin_amdgcn_global_load_async_to_lds_b128((as1_v4i*)g, (as3_v4i*)l, 0, 0);
#else
  *(uint4*)l = *(const uint4*)g;
#endif
}

static __device__ __forceinline__ void async_wait_all() {
#ifdef ASYNC_LDS
#if __has_builtin(__builtin_amdgcn_s_wait_asynccnt)
  __builtin_amdgcn_s_wait_asynccnt(0);
#else
  asm volatile("s_wait_asynccnt 0x0" ::: "memory");
#endif
#endif
}

// ---------------------------------------------------------------------------
// Weight cast + transpose:  W[K][N] f32  ->  Wt[N][K] bf16   (LDS tile)
// ---------------------------------------------------------------------------
__global__ __launch_bounds__(256) void wcast_t(const float* __restrict__ W,
                                               __bf16* __restrict__ Wt,
                                               int K, int N) {
  __shared__ float tile[32][33];
  const int tx = threadIdx.x;   // 0..31
  const int ty = threadIdx.y;   // 0..7
  const int n0 = blockIdx.x * 32;
  const int k0 = blockIdx.y * 32;
#pragma unroll
  for (int i = 0; i < 4; ++i)
    tile[ty + i * 8][tx] = W[(size_t)(k0 + ty + i * 8) * N + n0 + tx];
  __syncthreads();
#pragma unroll
  for (int i = 0; i < 4; ++i)
    Wt[(size_t)(n0 + ty + i * 8) * K + k0 + tx] = (__bf16)tile[tx][ty + i * 8];
}

// ---------------------------------------------------------------------------
// LayerNorm over C=1024, one block (256 threads) per row, bf16 out.
// ---------------------------------------------------------------------------
__global__ __launch_bounds__(256) void ln_kernel(const float* __restrict__ x,
                                                 const float* __restrict__ g,
                                                 const float* __restrict__ b,
                                                 __bf16* __restrict__ out) {
  __shared__ float red[8];
  __shared__ float red2[8];
  const int row  = blockIdx.x;
  const int wv   = threadIdx.x >> 5;
  const int lane = threadIdx.x & 31;

  const float4 v = ((const float4*)(x + (size_t)row * C_))[threadIdx.x];
  float s = v.x + v.y + v.z + v.w;
  s += __shfl_xor(s, 1, 32);  s += __shfl_xor(s, 2, 32);
  s += __shfl_xor(s, 4, 32);  s += __shfl_xor(s, 8, 32);
  s += __shfl_xor(s, 16, 32);
  if (!lane) red[wv] = s;
  __syncthreads();
  float tot = 0.0f;
#pragma unroll
  for (int i = 0; i < 8; ++i) tot += red[i];
  const float mu = tot * (1.0f / 1024.0f);

  const float dx = v.x - mu, dy = v.y - mu, dz = v.z - mu, dw = v.w - mu;
  float s2 = dx * dx + dy * dy + dz * dz + dw * dw;
  s2 += __shfl_xor(s2, 1, 32);  s2 += __shfl_xor(s2, 2, 32);
  s2 += __shfl_xor(s2, 4, 32);  s2 += __shfl_xor(s2, 8, 32);
  s2 += __shfl_xor(s2, 16, 32);
  if (!lane) red2[wv] = s2;
  __syncthreads();
  float var = 0.0f;
#pragma unroll
  for (int i = 0; i < 8; ++i) var += red2[i];
  var *= (1.0f / 1024.0f);
  const float rstd = rsqrtf(var + 1e-5f);

  const int i0 = threadIdx.x * 4;
  const float4 gg = ((const float4*)g)[threadIdx.x];
  const float4 bb = ((const float4*)b)[threadIdx.x];
  __bf16* op = out + (size_t)row * C_ + i0;
  op[0] = (__bf16)(dx * rstd * gg.x + bb.x);
  op[1] = (__bf16)(dy * rstd * gg.y + bb.y);
  op[2] = (__bf16)(dz * rstd * gg.z + bb.z);
  op[3] = (__bf16)(dw * rstd * gg.w + bb.w);
}

// ---------------------------------------------------------------------------
// WMMA GEMM:  C[M][N] = A_bf16[M][K] @ Wt_bf16[N][K]^T + bias, with epilogues.
//   MODE 0: QKV scatter (q scaled by 1/8, v stored transposed [h][d][t])
//   MODE 1: + resid (f32) -> outF f32
//   MODE 2: exact GELU -> outB bf16
// Block = 8 waves stacked in M (256x64 tile); wave computes 32x64.
// B slice (64x32 per K-step) staged in LDS via async global->LDS copies,
// double-buffered; each of 256 threads issues one b128 async load per step.
// ---------------------------------------------------------------------------
template <int MODE>
__global__ __launch_bounds__(256) void gemm_wmma(
    const __bf16* __restrict__ A, const __bf16* __restrict__ Wt,
    const float* __restrict__ bias, const float* __restrict__ resid,
    float* __restrict__ outF, __bf16* __restrict__ outB,
    __bf16* __restrict__ qb, __bf16* __restrict__ kb, __bf16* __restrict__ vT,
    int M, int N, int K) {
  __shared__ __align__(16) __bf16 bsh[2][64 * 32];   // 2 x 4 KB
  const int tid   = threadIdx.x;
  const int lane  = tid & 31;
  const int wv    = tid >> 5;
  const int tileM = blockIdx.y * 256 + wv * 32;
  const int tileN = blockIdx.x * 64;
  const int mRow  = lane & 15;            // A row / B column within tile
  const int kHalf = (lane >> 4) * 8;      // A-fragment K split
  const int kB16  = (lane >> 4) * 16;     // unused; B fragment also uses kHalf
  (void)M;

  // staging map: thread -> (column, k-quarter) of the 64x32 B slice
  const int sCol = tid >> 2;              // 0..63
  const int sK   = (tid & 3) * 8;         // 0,8,16,24
  const __bf16* bsrc = Wt + (size_t)(tileN + sCol) * K + sK;
  __bf16* bdst[2] = { &bsh[0][sCol * 32 + sK], &bsh[1][sCol * 32 + sK] };

  const __bf16* arow0 = A + (size_t)(tileM + mRow) * K;
  const __bf16* arow1 = A + (size_t)(tileM + 16 + mRow) * K;

  v8f acc[2][4];
#pragma unroll
  for (int rr = 0; rr < 2; ++rr)
#pragma unroll
    for (int t = 0; t < 4; ++t)
#pragma unroll
      for (int i = 0; i < 8; ++i) acc[rr][t][i] = 0.0f;

  // preload first B slice
  async_copy_b128(bsrc, bdst[0]);
  async_wait_all();
  __syncthreads();

  int buf = 0;
  for (int k0 = 0; k0 < K; k0 += 32) {
    if (k0 + 32 < K)                       // issue next slice into other buffer
      async_copy_b128(bsrc + k0 + 32, bdst[buf ^ 1]);

    Frag a0, a1;
    a0.u[0] = *(const uint4*)(arow0 + k0 + kHalf);
    a0.u[1] = *(const uint4*)(arow0 + k0 + kHalf + 16);
    a1.u[0] = *(const uint4*)(arow1 + k0 + kHalf);
    a1.u[1] = *(const uint4*)(arow1 + k0 + kHalf + 16);
    __builtin_prefetch(arow0 + k0 + 256, 0, 1);    // global_prefetch_b8
    __builtin_prefetch(arow1 + k0 + 256, 0, 1);

    const __bf16* bl = &bsh[buf][0];
#pragma unroll
    for (int t = 0; t < 4; ++t) {
      const __bf16* bp = bl + (t * 16 + mRow) * 32 + kHalf;
      Frag b;
      b.u[0] = *(const uint4*)bp;
      b.u[1] = *(const uint4*)(bp + 16);
      acc[0][t] = wmma_bf16(a0, b, acc[0][t]);
      acc[1][t] = wmma_bf16(a1, b, acc[1][t]);
    }

    async_wait_all();     // next slice landed
    __syncthreads();      // all waves done reading `buf`; next slice visible
    buf ^= 1;
  }

  const int rowOff = (lane >> 4) << 3;  // C/D layout: lanes 16-31 hold M+8
#pragma unroll
  for (int rr = 0; rr < 2; ++rr) {
#pragma unroll
    for (int t = 0; t < 4; ++t) {
      const int col = tileN + t * 16 + (lane & 15);
      const float bv = bias[col];
#pragma unroll
      for (int r = 0; r < 8; ++r) {
        const int row = tileM + rr * 16 + r + rowOff;
        const float v = acc[rr][t][r] + bv;
        if (MODE == 0) {
          const int sec = col >> 10;  // 0=q 1=k 2=v  (uniform per tile)
          const int cc  = col & 1023;
          const int h   = cc >> 6;
          const int d   = cc & 63;
          if (sec == 0)      qb[((size_t)h * T_ + row) * D_ + d] = (__bf16)(v * 0.125f);
          else if (sec == 1) kb[((size_t)h * T_ + row) * D_ + d] = (__bf16)v;
          else               vT[((size_t)h * D_ + d) * T_ + row] = (__bf16)v;
        } else if (MODE == 1) {
          outF[(size_t)row * N + col] = v + resid[(size_t)row * N + col];
        } else {  // MODE 2: exact GELU
          const float gl = 0.5f * v * (1.0f + erff(v * 0.70710678118654752f));
          outB[(size_t)row * N + col] = (__bf16)gl;
        }
      }
    }
  }
}

// ---------------------------------------------------------------------------
// Flash attention: one wave per (head, 32-row query block).
// Two 16-row Q subtiles share every K/V fragment (2x reuse).
// S = Q K^T (WMMA), causal mask, online softmax, O += P V (WMMA).
// P converted C-layout -> A-layout via wave-private LDS staging.
// ---------------------------------------------------------------------------
__global__ __launch_bounds__(256) void attn_kernel(const __bf16* __restrict__ qbuf,
                                                   const __bf16* __restrict__ kbuf,
                                                   const __bf16* __restrict__ vT,
                                                   __bf16* __restrict__ attn) {
  __shared__ __align__(16) __bf16 plds[8][32 * 64];   // 4 KB per wave
  const int lane   = threadIdx.x & 31;
  const int wv     = threadIdx.x >> 5;
  const int gid    = blockIdx.x * 8 + wv;      // 2048 waves total
  const int h      = gid >> 7;                 // / 128
  const int qblk   = gid & 127;
  const int q0     = qblk * 32;
  const int mRow   = lane & 15;
  const int kHalf  = (lane >> 4) * 8;
  const int kB16   = (lane >> 4) * 16;
  const int rowOff = (lane >> 4) << 3;
  const int nCol   = lane & 15;

  // Q fragments: 2 row-subtiles x 2 K-chunks, resident for the whole loop.
  Frag qf[2][2];
#pragma unroll
  for (int rr = 0; rr < 2; ++rr) {
    const __bf16* qp = qbuf + ((size_t)h * T_ + q0 + rr * 16 + mRow) * D_;
#pragma unroll
    for (int c = 0; c < 2; ++c) {
      qf[rr][c].u[0] = *(const uint4*)(qp + c * 32 + kHalf);
      qf[rr][c].u[1] = *(const uint4*)(qp + c * 32 + kHalf + 16);
    }
  }

  v8f o[2][4];
  float mrow[2][8], lrow[2][8];
#pragma unroll
  for (int rr = 0; rr < 2; ++rr) {
#pragma unroll
    for (int t = 0; t < 4; ++t)
#pragma unroll
      for (int i = 0; i < 8; ++i) o[rr][t][i] = 0.0f;
#pragma unroll
    for (int r = 0; r < 8; ++r) { mrow[rr][r] = -__builtin_inff(); lrow[rr][r] = 0.0f; }
  }

  const int kbmax = (q0 + 31) >> 6;
  for (int blk = 0; blk <= kbmax; ++blk) {
    const int s0 = blk * 64;

    // ---- S = Q K^T over two 16x64 score tiles ----------------------------
    v8f sacc[2][4];
#pragma unroll
    for (int rr = 0; rr < 2; ++rr)
#pragma unroll
      for (int t = 0; t < 4; ++t)
#pragma unroll
        for (int i = 0; i < 8; ++i) sacc[rr][t][i] = 0.0f;
#pragma unroll
    for (int t = 0; t < 4; ++t) {
      const __bf16* kp = kbuf + ((size_t)h * T_ + s0 + t * 16 + nCol) * D_ + kHalf;
#pragma unroll
      for (int c = 0; c < 2; ++c) {
        Frag kf;
        kf.u[0] = *(const uint4*)(kp + c * 32);
        kf.u[1] = *(const uint4*)(kp + c * 32 + 16);
        sacc[0][t] = wmma_bf16(qf[0][c], kf, sacc[0][t]);
        sacc[1][t] = wmma_bf16(qf[1][c], kf, sacc[1][t]);
      }
    }

    // ---- causal mask ------------------------------------------------------
    if (s0 + 63 > q0) {
#pragma unroll
      for (int rr = 0; rr < 2; ++rr)
#pragma unroll
        for (int t = 0; t < 4; ++t) {
          const int s = s0 + t * 16 + nCol;
#pragma unroll
          for (int r = 0; r < 8; ++r) {
            const int row = q0 + rr * 16 + r + rowOff;
            if (s > row) sacc[rr][t][r] = -__builtin_inff();
          }
        }
    }

    // ---- online softmax (row reductions across half-wave lanes) ----------
#pragma unroll
    for (int rr = 0; rr < 2; ++rr) {
      float mnew[8];
#pragma unroll
      for (int r = 0; r < 8; ++r) {
        float v = fmaxf(fmaxf(sacc[rr][0][r], sacc[rr][1][r]),
                        fmaxf(sacc[rr][2][r], sacc[rr][3][r]));
        v = fmaxf(v, __shfl_xor(v, 1, 32));
        v = fmaxf(v, __shfl_xor(v, 2, 32));
        v = fmaxf(v, __shfl_xor(v, 4, 32));
        v = fmaxf(v, __shfl_xor(v, 8, 32));
        mnew[r] = fmaxf(mrow[rr][r], v);
      }
#pragma unroll
      for (int r = 0; r < 8; ++r) {
        const float alpha = (mrow[rr][r] == -__builtin_inff())
                                ? 0.0f : __expf(mrow[rr][r] - mnew[r]);
        mrow[rr][r] = mnew[r];
        float rs = 0.0f;
#pragma unroll
        for (int t = 0; t < 4; ++t) {
          const float p = __expf(sacc[rr][t][r] - mnew[r]);
          sacc[rr][t][r] = p;
          rs += p;
        }
        rs += __shfl_xor(rs, 1, 32);
        rs += __shfl_xor(rs, 2, 32);
        rs += __shfl_xor(rs, 4, 32);
        rs += __shfl_xor(rs, 8, 32);
        lrow[rr][r] = lrow[rr][r] * alpha + rs;
#pragma unroll
        for (int t = 0; t < 4; ++t) o[rr][t][r] *= alpha;
      }
    }

    // ---- stage P (C-layout) into LDS as bf16, reload as A-fragments ------
    __bf16* pl = &plds[wv][0];
#pragma unroll
    for (int rr = 0; rr < 2; ++rr)
#pragma unroll
      for (int t = 0; t < 4; ++t)
#pragma unroll
        for (int r = 0; r < 8; ++r)
          pl[(rr * 16 + r + rowOff) * 64 + t * 16 + nCol] = (__bf16)sacc[rr][t][r];
    // same-wave LDS ops are in-order; compiler inserts s_wait_dscnt for VGPRs

    Frag pf[2][2];
#pragma unroll
    for (int rr = 0; rr < 2; ++rr)
#pragma unroll
      for (int c = 0; c < 2; ++c) {
        const __bf16* pp = pl + (rr * 16 + mRow) * 64 + c * 32 + kHalf;
        pf[rr][c].u[0] = *(const uint4*)pp;
        pf[rr][c].u[1] = *(const uint4*)(pp + 16);
      }

    // ---- O += P V  (V stored transposed [h][d][t], contiguous along s) ---
#pragma unroll
    for (int t = 0; t < 4; ++t) {
      const __bf16* vp = vT + ((size_t)h * D_ + t * 16 + nCol) * T_ + s0 + kHalf;
#pragma unroll
      for (int c = 0; c < 2; ++c) {
        Frag vf;
        vf.u[0] = *(const uint4*)(vp + c * 32);
        vf.u[1] = *(const uint4*)(vp + c * 32 + 16);
        o[0][t] = wmma_bf16(pf[0][c], vf, o[0][t]);
        o[1][t] = wmma_bf16(pf[1][c], vf, o[1][t]);
      }
    }
  }

  // ---- normalize and write attn output [T][C] bf16 -----------------------
#pragma unroll
  for (int rr = 0; rr < 2; ++rr) {
    float invl[8];
#pragma unroll
    for (int r = 0; r < 8; ++r) invl[r] = 1.0f / lrow[rr][r];
#pragma unroll
    for (int t = 0; t < 4; ++t)
#pragma unroll
      for (int r = 0; r < 8; ++r)
        attn[(size_t)(q0 + rr * 16 + r + rowOff) * C_ + h * D_ + t * 16 + nCol] =
            (__bf16)(o[rr][t][r] * invl[r]);
  }
}

// ---------------------------------------------------------------------------
// Host launch.  d_in order: x, ln1_g, ln1_b, w_qkv, b_qkv, w_out, b_out,
//                           ln2_g, ln2_b, w_fc1, b_fc1, w_fc2, b_fc2
// ---------------------------------------------------------------------------
extern "C" void kernel_launch(void* const* d_in, const int* in_sizes, int n_in,
                              void* d_out, int out_size, void* d_ws, size_t ws_size,
                              hipStream_t stream) {
  (void)in_sizes; (void)n_in; (void)out_size; (void)ws_size;
  const float* x     = (const float*)d_in[0];
  const float* ln1_g = (const float*)d_in[1];
  const float* ln1_b = (const float*)d_in[2];
  const float* w_qkv = (const float*)d_in[3];
  const float* b_qkv = (const float*)d_in[4];
  const float* w_out = (const float*)d_in[5];
  const float* b_out = (const float*)d_in[6];
  const float* ln2_g = (const float*)d_in[7];
  const float* ln2_b = (const float*)d_in[8];
  const float* w_fc1 = (const float*)d_in[9];
  const float* b_fc1 = (const float*)d_in[10];
  const float* w_fc2 = (const float*)d_in[11];
  const float* b_fc2 = (const float*)d_in[12];
  float* out = (float*)d_out;

  char* ws = (char*)d_ws;
  const size_t MB = 1ull << 20;
  __bf16* xn1   = (__bf16*)(ws + 0 * MB);    //  8 MB  [T][C]
  __bf16* wqkvT = (__bf16*)(ws + 8 * MB);    //  6 MB  [3C][C]
  __bf16* woutT = (__bf16*)(ws + 14 * MB);   //  2 MB  [C][C]
  __bf16* wfc1T = (__bf16*)(ws + 16 * MB);   //  8 MB  [4C][C]
  __bf16* wfc2T = (__bf16*)(ws + 24 * MB);   //  8 MB  [C][4C]
  __bf16* qb    = (__bf16*)(ws + 32 * MB);   //  8 MB  [H][T][D]
  __bf16* kb    = (__bf16*)(ws + 40 * MB);   //  8 MB  [H][T][D]
  __bf16* vT    = (__bf16*)(ws + 48 * MB);   //  8 MB  [H][D][T]
  __bf16* attn  = (__bf16*)(ws + 56 * MB);   //  8 MB  [T][C]
  float*  x1    = (float*) (ws + 64 * MB);   // 16 MB  [T][C]
  __bf16* xn2   = (__bf16*)(ws + 80 * MB);   //  8 MB  [T][C]
  __bf16* hbuf  = (__bf16*)(ws + 88 * MB);   // 32 MB  [T][4C]

  const dim3 tb(32, 8);
  wcast_t<<<dim3(3072 / 32, 1024 / 32), tb, 0, stream>>>(w_qkv, wqkvT, 1024, 3072);
  wcast_t<<<dim3(1024 / 32, 1024 / 32), tb, 0, stream>>>(w_out, woutT, 1024, 1024);
  wcast_t<<<dim3(4096 / 32, 1024 / 32), tb, 0, stream>>>(w_fc1, wfc1T, 1024, 4096);
  wcast_t<<<dim3(1024 / 32, 4096 / 32), tb, 0, stream>>>(w_fc2, wfc2T, 4096, 1024);

  ln_kernel<<<T_, 256, 0, stream>>>(x, ln1_g, ln1_b, xn1);

  gemm_wmma<0><<<dim3(3072 / 64, T_ / 256), 256, 0, stream>>>(
      xn1, wqkvT, b_qkv, nullptr, nullptr, nullptr, qb, kb, vT, T_, 3072, 1024);

  attn_kernel<<<(H_ * (T_ / 32)) / 8, 256, 0, stream>>>(qb, kb, vT, attn);

  gemm_wmma<1><<<dim3(1024 / 64, T_ / 256), 256, 0, stream>>>(
      attn, woutT, b_out, x, x1, nullptr, nullptr, nullptr, nullptr, T_, 1024, 1024);

  ln_kernel<<<T_, 256, 0, stream>>>(x1, ln2_g, ln2_b, xn2);

  gemm_wmma<2><<<dim3(4096 / 64, T_ / 256), 256, 0, stream>>>(
      xn2, wfc1T, b_fc1, nullptr, nullptr, hbuf, nullptr, nullptr, nullptr, T_, 4096, 1024);

  gemm_wmma<1><<<dim3(1024 / 64, T_ / 256), 256, 0, stream>>>(
      hbuf, wfc2T, b_fc2, x1, out, nullptr, nullptr, nullptr, nullptr, T_, 1024, 4096);
}
